// GRUPlant_7164005450351
// MI455X (gfx1250) — matrix operand order repaired and checked by reading, not compiled, not run on hardware
//
#include <hip/hip_runtime.h>

typedef __attribute__((ext_vector_type(16))) _Float16 v16h;
typedef __attribute__((ext_vector_type(8)))  float    v8f;

#define B_TOT 256
#define T_IN  512
#define T_TGT 256
#define N_IN  64
#define N_PV  16
#define HDIM  256
#define EMB   32
#define ROWS  16          // batch rows per workgroup (one M tile)
#define NWG   (B_TOT / ROWS)
#define THREADS 512       // 16 waves of 32

// Workspace offsets (in _Float16 elements) for f16 B-fragment weights.
#define O_E0I 0u          // enc Wih0: 48*3*512
#define O_E0H 73728u      // enc Whh0: 48*8*512
#define O_E1I 270336u     // enc Wih1
#define O_E1H 466944u     // enc Whh1
#define O_D0I 663552u     // dec Wih0 (K=80 padded to 96)
#define O_D0H 737280u     // dec Whh0
#define O_D1I 933888u     // dec Wih1
#define O_D1H 1130496u    // dec Whh1
#define O_FC1 1327104u    // fc1: 8 N-tiles * 8 chunks * 512
#define O_FC2 1359872u    // fc2: 1 N-tile * 4 chunks * 512

// ---------- fragment layout helpers (wave32 WMMA f16, 16x16x32) ----------
// A-matrix (16xK) LDS layout, per 32-K chunk: slot[(chunk*32+lane)*16 + e]
//   lane = m + 16*hi,  K = chunk*32 + e + 8*(e/8) + 8*hi   (ISA 7.12.2)
__device__ __forceinline__ int fragSlot(int m, int k) {
    int c  = k >> 5;
    int kk = k & 31;
    int e  = (kk & 7) + 8 * (kk >> 4);
    int hi = (kk >> 3) & 1;
    int lane = m + 16 * hi;
    return ((c * 32 + lane) << 4) + e;
}

// B-matrix (Kx16): lane -> N = lane%16, K = chunk*32 + (e%8) + 8*(lane/16) + 16*(e/8)
// Weights pre-swizzled to this layout so each lane reads 32 contiguous bytes.

__device__ __forceinline__ v8f gemm_frag(const _Float16* aLds,
                                         const _Float16* bGlb,
                                         int chunks, int lane, v8f acc) {
#pragma unroll
    for (int c = 0; c < chunks; ++c) {
        v16h a = *(const v16h*)(aLds + ((c * 32 + lane) << 4));
        v16h b = *(const v16h*)(bGlb + ((c * 32 + lane) << 4));
        acc = __builtin_amdgcn_wmma_f32_16x16x32_f16(
            false, a, false, b, (short)0, acc, false, false);
    }
    return acc;
}

// Convert fp32 weight W[nOut][Kin] -> f16 B-fragment layout (K padded to kc*32).
__device__ __forceinline__ void convW(const float* __restrict__ W,
                                      _Float16* __restrict__ dst,
                                      int Kin, int ntiles, int kc, int tid) {
    int total = ntiles * kc * 512;
    for (int idx = tid; idx < total; idx += THREADS) {
        int e    = idx & 15;
        int lane = (idx >> 4) & 31;
        int c    = (idx >> 9) % kc;
        int nt   = idx / (kc * 512);
        int n    = nt * 16 + (lane & 15);
        int k    = c * 32 + (e & 7) + 8 * (lane >> 4) + 16 * (e >> 3);
        float v  = (k < Kin) ? W[n * Kin + k] : 0.0f;
        dst[idx] = (_Float16)v;
    }
}

__device__ __forceinline__ float sigmoidf_(float x) {
    return 1.0f / (1.0f + __expf(-x));
}

// Bias table entry: [0..511] = bih+bhh (r|z), [512..767] = bih_n, [768..1023] = bhh_n
__device__ __forceinline__ float biasVal(const float* bi, const float* bh, int i) {
    if (i < 512) return bi[i] + bh[i];
    if (i < 768) return bi[i];        // bih[512 + (i-512)] == bih[i]
    return bh[i - 256];               // bhh[512 + (i-768)] == bhh[i-256]
}

// One GRU layer step for 16 batch rows. Wave w owns hidden columns [16w,16w+16):
// tiles w (r), 16+w (z), 32+w (n). Whole nonlinearity stays in registers.
__device__ __forceinline__ void gru_layer(const _Float16* __restrict__ aX, int kcX,
                                          const _Float16* __restrict__ Wi,
                                          const _Float16* __restrict__ hCur,
                                          const _Float16* __restrict__ Wh,
                                          _Float16* __restrict__ hNxt,
                                          const float* __restrict__ lb,
                                          int wave, int lane) {
    v8f ar = {};   // r gate: gi + gh accumulated together
    ar = gemm_frag(aX,   Wi + (size_t)wave * kcX * 512, kcX, lane, ar);
    ar = gemm_frag(hCur, Wh + (size_t)wave * 8 * 512,   8,   lane, ar);
    v8f az = {};   // z gate
    az = gemm_frag(aX,   Wi + (size_t)(16 + wave) * kcX * 512, kcX, lane, az);
    az = gemm_frag(hCur, Wh + (size_t)(16 + wave) * 8 * 512,   8,   lane, az);
    v8f ani = {};  // n gate, input part
    ani = gemm_frag(aX,  Wi + (size_t)(32 + wave) * kcX * 512, kcX, lane, ani);
    v8f anh = {};  // n gate, hidden part
    anh = gemm_frag(hCur, Wh + (size_t)(32 + wave) * 8 * 512,  8,   lane, anh);

    int nl = lane & 15, mb = (lane >> 4) * 8;
    int col = wave * 16 + nl;                 // hidden column in [0,256)
    float brz_r = lb[col];
    float brz_z = lb[256 + col];
    float bin   = lb[512 + col];
    float bhn   = lb[768 + col];
    int slot0 = fragSlot(mb, col);            // slot(mb+v,col) = slot0 + 16*v
#pragma unroll
    for (int v = 0; v < 8; ++v) {
        float r = sigmoidf_(ar[v] + brz_r);
        float z = sigmoidf_(az[v] + brz_z);
        float n = tanhf(ani[v] + bin + r * (anh[v] + bhn));
        float hp = (float)hCur[slot0 + (v << 4)];
        hNxt[slot0 + (v << 4)] = (_Float16)((1.0f - z) * n + z * hp);
    }
}

__global__ __launch_bounds__(THREADS, 1) void grup_kernel(
    const float* __restrict__ x_cv, const float* __restrict__ x_tgt,
    const float* __restrict__ pv_init, const int* __restrict__ scen,
    const float* __restrict__ emb,
    const float* eW0i, const float* eW0h, const float* eb0i, const float* eb0h,
    const float* eW1i, const float* eW1h, const float* eb1i, const float* eb1h,
    const float* dW0i, const float* dW0h, const float* db0i, const float* db0h,
    const float* dW1i, const float* dW1h, const float* db1i, const float* db1h,
    const float* fc1W, const float* fc1b, const float* fc2W, const float* fc2b,
    float* __restrict__ out, _Float16* __restrict__ wbuf) {

    __shared__ _Float16 xfrag[3 * 512];       // K=96 input frags (x | emb/pv)
    __shared__ _Float16 h0A[8 * 512], h0B[8 * 512];   // h0 ping-pong A-frags
    __shared__ _Float16 h1A[8 * 512], h1B[8 * 512];   // h1 ping-pong A-frags
    __shared__ _Float16 fcfrag[4 * 512];      // fc1 output (16x128) A-frags
    __shared__ float    lbias[4 * 1024];      // enc0|enc1|dec0|dec1 bias tables

    const int tid   = threadIdx.x;
    const int lane  = tid & 31;
    const int wave  = tid >> 5;
    const int bbase = blockIdx.x * ROWS;

    // ---- one-time weight conversion (every WG writes identical data) ----
    convW(eW0i, wbuf + O_E0I, 96, 48, 3, tid);
    convW(eW0h, wbuf + O_E0H, 256, 48, 8, tid);
    convW(eW1i, wbuf + O_E1I, 256, 48, 8, tid);
    convW(eW1h, wbuf + O_E1H, 256, 48, 8, tid);
    convW(dW0i, wbuf + O_D0I, 80, 48, 3, tid);   // pad K 80->96
    convW(dW0h, wbuf + O_D0H, 256, 48, 8, tid);
    convW(dW1i, wbuf + O_D1I, 256, 48, 8, tid);
    convW(dW1h, wbuf + O_D1H, 256, 48, 8, tid);
    convW(fc1W, wbuf + O_FC1, 256, 8, 8, tid);
    convW(fc2W, wbuf + O_FC2, 128, 1, 4, tid);

    // ---- bias tables ----
    for (int i = tid; i < 1024; i += THREADS) {
        lbias[0 * 1024 + i] = biasVal(eb0i, eb0h, i);
        lbias[1 * 1024 + i] = biasVal(eb1i, eb1h, i);
        lbias[2 * 1024 + i] = biasVal(db0i, db0h, i);
        lbias[3 * 1024 + i] = biasVal(db1i, db1h, i);
    }

    // ---- init state: h = 0 (parity 0 buffers), xfrag cols 64..95 = embedding ----
    for (int i = tid; i < 8 * 512; i += THREADS) { h0A[i] = (_Float16)0.0f; h1A[i] = (_Float16)0.0f; }
    for (int i = tid; i < ROWS * EMB; i += THREADS) {
        int m = i >> 5, j = i & 31;
        xfrag[fragSlot(m, 64 + j)] = (_Float16)emb[scen[bbase + m] * EMB + j];
    }
    __syncthreads();

    int pb = 0;  // ping-pong parity, carries across encoder -> decoder

    for (int phase = 0; phase < 2; ++phase) {
        const bool dec = (phase == 1);
        const int Tlen = dec ? T_TGT : T_IN;
        const float* xsrc = dec ? x_tgt : x_cv;
        const _Float16* W0i = wbuf + (dec ? O_D0I : O_E0I);
        const _Float16* W0h = wbuf + (dec ? O_D0H : O_E0H);
        const _Float16* W1i = wbuf + (dec ? O_D1I : O_E1I);
        const _Float16* W1h = wbuf + (dec ? O_D1H : O_E1H);
        const float* lb0 = lbias + (dec ? 2 : 0) * 1024;
        const float* lb1 = lbias + (dec ? 3 : 1) * 1024;

        if (dec) {
            // decoder init: xfrag cols 64..79 = pv_init, 80..95 = 0
            for (int i = tid; i < ROWS * 32; i += THREADS) {
                int m = i >> 5, j = i & 31;
                float v = (j < N_PV) ? pv_init[(bbase + m) * N_PV + j] : 0.0f;
                xfrag[fragSlot(m, 64 + j)] = (_Float16)v;
            }
            __syncthreads();
        }

        for (int t = 0; t < Tlen; ++t) {
            _Float16* h0c = pb ? h0B : h0A;
            _Float16* h0n = pb ? h0A : h0B;
            _Float16* h1c = pb ? h1B : h1A;
            _Float16* h1n = pb ? h1A : h1B;

            // 1) load x_t into xfrag cols 0..63
            for (int i = tid; i < ROWS * N_IN; i += THREADS) {
                int m = i >> 6, j = i & 63;
                float v = xsrc[((size_t)(bbase + m) * Tlen + t) * N_IN + j];
                xfrag[fragSlot(m, j)] = (_Float16)v;
            }
            __syncthreads();

            // 2) layer 0: reads xfrag + h0c, writes h0n (fully in-register combine)
            gru_layer(xfrag, 3, W0i, h0c, W0h, h0n, lb0, wave, lane);
            __syncthreads();

            // 3) layer 1: reads h0n + h1c, writes h1n
            gru_layer(h0n, 8, W1i, h1c, W1h, h1n, lb1, wave, lane);
            __syncthreads();

            // 4) decoder head: y = relu(h1@fc1^T+b)@fc2^T+b ; feed pv back
            if (dec) {
                if (wave < 8) {   // fc1: 8 N-tiles of 16, K=256
                    v8f acc = {};
                    acc = gemm_frag(h1n, wbuf + O_FC1 + (size_t)wave * 8 * 512, 8, lane, acc);
                    int nl = lane & 15, mb = (lane >> 4) * 8, col = wave * 16 + nl;
                    float bb = fc1b[col];
                    int slot0 = fragSlot(mb, col);
#pragma unroll
                    for (int v = 0; v < 8; ++v) {
                        float val = acc[v] + bb;
                        val = val > 0.0f ? val : 0.0f;
                        fcfrag[slot0 + (v << 4)] = (_Float16)val;
                    }
                }
                __syncthreads();
                if (wave == 0) {  // fc2: 1 N-tile, K=128
                    v8f acc = {};
                    acc = gemm_frag(fcfrag, wbuf + O_FC2, 4, lane, acc);
                    int nl = lane & 15, mb = (lane >> 4) * 8;
                    float bb = fc2b[nl];
                    int slot0 = fragSlot(mb, 64 + nl);
#pragma unroll
                    for (int v = 0; v < 8; ++v) {
                        int m = mb + v;
                        float y = acc[v] + bb;
                        out[((size_t)(bbase + m) * T_TGT + t) * N_PV + nl] = y;
                        xfrag[slot0 + (v << 4)] = (_Float16)y;   // pv feedback
                    }
                }
                __syncthreads();
            }
            pb ^= 1;
        }
    }
}

extern "C" void kernel_launch(void* const* d_in, const int* in_sizes, int n_in,
                              void* d_out, int out_size, void* d_ws, size_t ws_size,
                              hipStream_t stream) {
    const float* x_cv    = (const float*)d_in[0];
    const float* x_tgt   = (const float*)d_in[1];
    const float* pv_init = (const float*)d_in[2];
    const int*   scen    = (const int*)d_in[3];
    const float* emb     = (const float*)d_in[4];
    const float* eW0i = (const float*)d_in[5],  *eW0h = (const float*)d_in[6];
    const float* eb0i = (const float*)d_in[7],  *eb0h = (const float*)d_in[8];
    const float* eW1i = (const float*)d_in[9],  *eW1h = (const float*)d_in[10];
    const float* eb1i = (const float*)d_in[11], *eb1h = (const float*)d_in[12];
    const float* dW0i = (const float*)d_in[13], *dW0h = (const float*)d_in[14];
    const float* db0i = (const float*)d_in[15], *db0h = (const float*)d_in[16];
    const float* dW1i = (const float*)d_in[17], *dW1h = (const float*)d_in[18];
    const float* db1i = (const float*)d_in[19], *db1h = (const float*)d_in[20];
    const float* fc1W = (const float*)d_in[21], *fc1b = (const float*)d_in[22];
    const float* fc2W = (const float*)d_in[23], *fc2b = (const float*)d_in[24];

    grup_kernel<<<dim3(NWG), dim3(THREADS), 0, stream>>>(
        x_cv, x_tgt, pv_init, scen, emb,
        eW0i, eW0h, eb0i, eb0h, eW1i, eW1h, eb1i, eb1h,
        dW0i, dW0h, db0i, db0h, dW1i, dW1h, db1i, db1h,
        fc1W, fc1b, fc2W, fc2b,
        (float*)d_out, (_Float16*)d_ws);
}
